// NodeFormerConv_45621142618210
// MI455X (gfx1250) — matrix operand order repaired and checked by reading, not compiled, not run on hardware
//
#include <hip/hip_runtime.h>
#include <hip/hip_bf16.h>

typedef __attribute__((ext_vector_type(16))) __bf16 v16bf;
typedef __attribute__((ext_vector_type(8)))  __bf16 v8bf;
typedef __attribute__((ext_vector_type(8)))  float  v8f;

#define N_TOT 50000
#define MM 30
#define MP 32
#define KK 10
#define RATIO 0.18257418583505537f    /* 1/sqrt(30) */
#define NORMER 0.35355339059327373f   /* 64^-0.25 */

__device__ __forceinline__ __bf16 f2bf(float x){ return (__bf16)x; }

__device__ __forceinline__ v8f wmma_bf16(v16bf a, v16bf b, v8f c){
  return __builtin_amdgcn_wmma_f32_16x16x32_bf16(false, a, false, b, (short)0, c, false, false);
}

// A fragment 16x32 bf16 from row-major source (leading dim ld, elements).
// Per lane: two contiguous 8-element runs -> 2 x b128 loads.
__device__ __forceinline__ v16bf load_a_vec(const __bf16* base, int ld, int lane){
  const __bf16* p = base + (lane & 15) * ld + ((lane & 16) ? 16 : 0);
  v8bf lo = *(const v8bf*)(p);
  v8bf hi = *(const v8bf*)(p + 8);
  return __builtin_shufflevector(lo, hi, 0,1,2,3,4,5,6,7,8,9,10,11,12,13,14,15);
}

// B fragment 32x16 bf16 from a packed tile stored [col][K] (col-major, 512 elems).
// Per lane: 16 contiguous bf16 = one 32B read.
__device__ __forceinline__ v16bf load_b_packed(const __bf16* tile, int lane){
  return *(const v16bf*)(tile + (lane & 15)*32 + ((lane & 16) ? 16 : 0));
}

// ---------------- K0: weight/proj conversion (fragment-packed) + zeroing --
// wcat:  2 k-tiles x 48 col-tiles, each tile [16 col][32 K] bf16
// wo_b:  8 k-tiles x 4 col-tiles, same tile format
// projb: 2 k-tiles x 2 m-tiles (zero-padded m>=30), same tile format
__global__ __launch_bounds__(256) void k_prep(const float* __restrict__ Wq,
    const float* __restrict__ Wk, const float* __restrict__ Wv,
    const float* __restrict__ Wo, const float* __restrict__ proj,
    __bf16* __restrict__ wcat, __bf16* __restrict__ wo_b,
    float* __restrict__ kvs, float* __restrict__ kssum,
    __bf16* __restrict__ qp_bf_pad, __bf16* __restrict__ projb)
{
  int idx = blockIdx.x*256 + threadIdx.x;
  if (idx < 64*768){
    int kk = idx & 31;
    int t  = idx >> 5;
    int col = t & 15;
    int u  = t >> 4;
    int ct = u % 48, kt = u / 48;
    int r = kt*32 + kk;
    int c = ct*16 + col;
    float v;
    if (c < 256)      v = Wq[r*256 + c];
    else if (c < 512) v = Wk[r*256 + (c-256)];
    else              v = Wv[r*256 + (c-512)];
    wcat[idx] = f2bf(v);
    return;
  }
  idx -= 64*768;
  if (idx < 256*64){
    int kk = idx & 31;
    int t  = idx >> 5;
    int col = t & 15;
    int u  = t >> 4;           // 0..31
    int ct = u & 3, kt = u >> 2;
    wo_b[idx] = f2bf(Wo[(kt*32 + kk)*64 + ct*16 + col]);
    return;
  }
  idx -= 256*64;
  if (idx < 4*KK*MP*64){ kvs[idx] = 0.f; return; }
  idx -= 4*KK*MP*64;
  if (idx < 4*KK*MP){ kssum[idx] = 0.f; return; }
  idx -= 4*KK*MP;
  if (idx < 64*4*MP){ qp_bf_pad[idx] = f2bf(0.f); return; }  // zero qp_bf tail
  idx -= 64*4*MP;
  if (idx < 4*512){
    int kk = idx & 31;
    int col = (idx >> 5) & 15;
    int u = idx >> 9;          // 0..3
    int mt = u & 1, kt = u >> 1;
    int m = mt*16 + col, d = kt*32 + kk;
    projb[idx] = f2bf((m < MM) ? proj[m*64 + d] : 0.f);
  }
}

// ---- K1: fused QKV GEMM + WMMA phi-projection + feature epilogue ---------
__global__ __launch_bounds__(256) void k_qkv(const float* __restrict__ z,
    const __bf16* __restrict__ wcat, const __bf16* __restrict__ projb,
    const float* __restrict__ bq, const float* __restrict__ bk,
    const float* __restrict__ bv, const float* __restrict__ taup,
    float* __restrict__ vbuf, float* __restrict__ qp, __bf16* __restrict__ qp_bf,
    float* __restrict__ kd, float* __restrict__ diagk, float* __restrict__ pmax)
{
  __shared__ __align__(32) __bf16 zs[16][64];
  __shared__ __align__(32) __bf16 qsb[16][256];
  __shared__ __align__(32) __bf16 kshb[16][256];
  __shared__ float ddq_s[4][16][32];
  __shared__ float ddk_s[4][16][32];
  __shared__ float smax[64];

  const int tid = threadIdx.x, lane = tid & 31, wv = tid >> 5;
  const int n0 = blockIdx.x * 16;   // 3125 blocks: N divisible by 16

  {
    int r = tid >> 4, c4 = (tid & 15) * 4;     // 256 threads cover 16x64
    float4 zv = *(const float4*)(z + (size_t)(n0 + r)*64 + c4);
    zs[r][c4+0] = f2bf(zv.x); zs[r][c4+1] = f2bf(zv.y);
    zs[r][c4+2] = f2bf(zv.z); zs[r][c4+3] = f2bf(zv.w);
  }
  __syncthreads();

  const float s_qk = rsqrtf(*taup) * NORMER;

  v16bf a0 = load_a_vec(&zs[0][0], 64, lane);
  v16bf a1 = load_a_vec(&zs[0][0] + 32, 64, lane);

  const int col16 = lane & 15;
  const int rb = (lane & 16) ? 8 : 0;

  // ---- phase 1: P = z @ [Wq|Wk|Wv] ----
  for (int ct = wv; ct < 48; ct += 8){
    v8f c = {};
    v16bf b0 = load_b_packed(wcat + (size_t)ct*512, lane);
    c = wmma_bf16(a0, b0, c);
    v16bf b1 = load_b_packed(wcat + (size_t)(48 + ct)*512, lane);
    c = wmma_bf16(a1, b1, c);

    int colbase = ct * 16;
    int col = colbase + col16;
    if (colbase < 256){
      float bias = bq[col];
#pragma unroll
      for (int r = 0; r < 8; ++r) qsb[rb + r][col] = f2bf((c[r] + bias) * s_qk);
    } else if (colbase < 512){
      float bias = bk[col - 256];
#pragma unroll
      for (int r = 0; r < 8; ++r) kshb[rb + r][col - 256] = f2bf((c[r] + bias) * s_qk);
    } else {
      float bias = bv[col - 512];
#pragma unroll
      for (int r = 0; r < 8; ++r) vbuf[(size_t)(n0 + rb + r)*256 + (col - 512)] = c[r] + bias;
    }
  }
  __syncthreads();

  // ---- phase 1b: dd = data @ proj^T via WMMA (16 tiles: qk x head x mtile)
  for (int tt = wv; tt < 16; tt += 8){
    int qk = tt & 1;
    int u  = tt >> 1;
    int h = u & 3, mt = u >> 2;
    const __bf16* src = qk ? &kshb[0][0] : &qsb[0][0];
    v8f c = {};
    v16bf pa0 = load_a_vec(src + h*64, 256, lane);
    v16bf pb0 = load_b_packed(projb + (size_t)mt*512, lane);
    c = wmma_bf16(pa0, pb0, c);
    v16bf pa1 = load_a_vec(src + h*64 + 32, 256, lane);
    v16bf pb1 = load_b_packed(projb + (size_t)(2 + mt)*512, lane);
    c = wmma_bf16(pa1, pb1, c);
    float* dst = qk ? &ddk_s[h][0][0] : &ddq_s[h][0][0];
#pragma unroll
    for (int r = 0; r < 8; ++r) dst[(rb + r)*32 + mt*16 + col16] = c[r];
  }
  __syncthreads();

  // ---- phase 2: diag, max, exp epilogue ----
  if (tid < 64){
    int r = tid >> 2, h = tid & 3;
    int n = n0 + r;
    const __bf16* dq = &qsb[r][h*64];
    const __bf16* dk = &kshb[r][h*64];
    float diagq = 0.f, dgk = 0.f;
#pragma unroll 4
    for (int d = 0; d < 64; ++d){
      float aq = (float)dq[d], ak = (float)dk[d];
      diagq += aq*aq; dgk += ak*ak;
    }
    diagq *= 0.5f; dgk *= 0.5f;

    const float* ddq = &ddq_s[h][r][0];
    const float* ddk = &ddk_s[h][r][0];
    float mq = -3.4e38f, mk = -3.4e38f;
    for (int m = 0; m < MM; ++m){
      mq = fmaxf(mq, ddq[m]); mk = fmaxf(mk, ddk[m]);
    }
    float*  qpo = &qp[(size_t)(n*4 + h)*MP];
    __bf16* qbo = &qp_bf[(size_t)(n*4 + h)*MP];
    float*  kdo = &kd[(size_t)(n*4 + h)*MP];
    for (int m = 0; m < MM; ++m){
      float qv = RATIO * (__expf(ddq[m] - diagq - mq) + 1e-6f);
      qpo[m] = qv;
      qbo[m] = f2bf(qv);
      kdo[m] = ddk[m];
    }
    qpo[30] = 0.f; qpo[31] = 0.f;          // pad -> inert in WMMA contractions
    qbo[30] = f2bf(0.f); qbo[31] = f2bf(0.f);
    kdo[30] = -3.4e38f; kdo[31] = -3.4e38f;
    diagk[n*4 + h] = dgk;
    smax[tid] = mk;
  }
  __syncthreads();
  if (tid < 4){
    float m = -3.4e38f;
    for (int i = tid; i < 64; i += 4) m = fmaxf(m, smax[i]);
    pmax[blockIdx.x*4 + tid] = m;
  }
}

// ---------------- K2: global per-head max over N --------------------------
__global__ __launch_bounds__(256) void k_hmax(const float* __restrict__ pmax,
                                              float* __restrict__ hmax, int nb)
{
  __shared__ float sm[256];
  const int h = blockIdx.x, tid = threadIdx.x;
  float m = -3.4e38f;
  for (int i = tid; i < nb; i += 256) m = fmaxf(m, pmax[i*4 + h]);
  sm[tid] = m; __syncthreads();
  for (int s = 128; s > 0; s >>= 1){
    if (tid < s) sm[tid] = fmaxf(sm[tid], sm[tid + s]);
    __syncthreads();
  }
  if (tid == 0) hmax[h] = sm[0];
}

// ---------------- K3: kp = ratio*(exp(dd - diag - hmax)+eps), in place ----
__global__ __launch_bounds__(256) void k_kp(float* __restrict__ kd,
    const float* __restrict__ diagk, const float* __restrict__ hmax)
{
  int idx = blockIdx.x*256 + threadIdx.x;
  if (idx >= N_TOT*4*MP) return;
  int m = idx & 31;
  int nh = idx >> 5;
  int h = nh & 3;
  float v = 0.f;
  if (m < MM) v = RATIO * (__expf(kd[idx] - diagk[nh] - hmax[h]) + 1e-6f);
  kd[idx] = v;
}

// -------- K4: kvs[h,k,m,d] += sum_n kp*g*v (WMMA over N), + ks_sum --------
__global__ __launch_bounds__(256) void k_kvs(const float* __restrict__ kp,
    const float* __restrict__ vbuf, const float* __restrict__ gum,
    const float* __restrict__ taup, float* __restrict__ kvs,
    float* __restrict__ kssum)
{
  __shared__ __align__(32) __bf16 a_s[320][32];   // A^T rows=(k,m), K=n-chunk
  __shared__ __align__(32) __bf16 b_sp[64*32];    // packed: [d][j] col-major tiles
  __shared__ float kp_s[32][32];
  __shared__ float g_s[32][KK];
  __shared__ float ks_local[320];

  const int tid = threadIdx.x, lane = tid & 31, wv = tid >> 5;
  const int h = blockIdx.y;
  const int nbase = blockIdx.x * 512;
  const float inv_tau = 1.0f / (*taup);

  for (int i = tid; i < 320; i += 256) ks_local[i] = 0.f;

  v8f zero = {};
  v8f acc[10];
#pragma unroll
  for (int i = 0; i < 10; ++i) acc[i] = zero;

  for (int s = 0; s < 16; ++s){
    int nb = nbase + s*32;
    {   // kp tile: 32 rows x 32 m via float4
      int j = tid >> 3, m4 = (tid & 7) * 4;
      int n = nb + j;
      float4 kv = (n < N_TOT) ? *(const float4*)(kp + (size_t)(n*4 + h)*MP + m4)
                              : make_float4(0.f, 0.f, 0.f, 0.f);
      kp_s[j][m4+0] = kv.x; kp_s[j][m4+1] = kv.y;
      kp_s[j][m4+2] = kv.z; kp_s[j][m4+3] = kv.w;
    }
    for (int i = tid; i < 32*KK; i += 256){
      int j = i / KK, k = i % KK;
      int n = nb + j;
      g_s[j][k] = (n < N_TOT) ? __expf(gum[(size_t)(n*4 + h)*KK + k] * inv_tau) : 0.f;
    }
    for (int i = tid; i < 512; i += 256){   // v tile, transposed into [d][j]
      int j = i >> 4, d4 = (i & 15) * 4;
      int n = nb + j;
      float4 vv = (n < N_TOT) ? *(const float4*)(vbuf + (size_t)n*256 + h*64 + d4)
                              : make_float4(0.f, 0.f, 0.f, 0.f);
      b_sp[(d4+0)*32 + j] = f2bf(vv.x);
      b_sp[(d4+1)*32 + j] = f2bf(vv.y);
      b_sp[(d4+2)*32 + j] = f2bf(vv.z);
      b_sp[(d4+3)*32 + j] = f2bf(vv.w);
    }
    __syncthreads();

    for (int km = tid; km < 320; km += 256){
      int k = km >> 5, m = km & 31;
      float srow = 0.f;
      __bf16* arow = &a_s[km][0];
#pragma unroll
      for (int jb = 0; jb < 4; ++jb){         // 4 x (8-wide build + b128 store)
        v8bf chunk;
#pragma unroll
        for (int j = 0; j < 8; ++j){
          float val = kp_s[jb*8 + j][m] * g_s[jb*8 + j][k];
          chunk[j] = f2bf(val);
          srow += val;
        }
        *(v8bf*)(arow + jb*8) = chunk;
      }
      ks_local[km] += srow;    // unique km per thread: no atomics needed
    }
    __syncthreads();

#pragma unroll
    for (int ti = 0; ti < 10; ++ti){
      int t = wv*10 + ti;
      int rt = t >> 2, ctile = t & 3;
      v16bf a = load_a_vec(&a_s[rt*16][0], 32, lane);
      v16bf b = load_b_packed(&b_sp[ctile*16*32], lane);
      acc[ti] = wmma_bf16(a, b, acc[ti]);
    }
    __syncthreads();
  }

  const int col16 = lane & 15;
  const int rb = (lane & 16) ? 8 : 0;
#pragma unroll
  for (int ti = 0; ti < 10; ++ti){
    int t = wv*10 + ti;
    int rt = t >> 2, ctile = t & 3;
    int col = ctile*16 + col16;
#pragma unroll
    for (int r = 0; r < 8; ++r){
      int km = rt*16 + rb + r;
      atomicAdd(&kvs[h*(KK*MP*64) + km*64 + col], acc[ti][r]);
    }
  }
  for (int i = tid; i < 320; i += 256) atomicAdd(&kssum[h*320 + i], ks_local[i]);
}

// ------- K4b: repack kvs (fp32 [h,k][m][d]) -> bf16 tiles [h,k][dt][col][m]
__global__ __launch_bounds__(256) void k_cvt(const float* __restrict__ kvs,
                                             __bf16* __restrict__ kvs_bf)
{
  int idx = blockIdx.x*256 + threadIdx.x;
  if (idx >= 4*KK*4*16*32) return;               // 81920
  int m   = idx & 31;
  int col = (idx >> 5) & 15;
  int dt  = (idx >> 9) & 3;
  int hk  = idx >> 11;                           // 0..39
  kvs_bf[idx] = f2bf(kvs[(size_t)hk*2048 + m*64 + dt*16 + col]);
}

// ---------------- K5: z_den[n,h,k] = sum_m qp * ks_sum --------------------
__global__ __launch_bounds__(256) void k_zden(const float* __restrict__ qp,
    const float* __restrict__ kssum, float* __restrict__ zden)
{
  int idx = blockIdx.x*256 + threadIdx.x;
  if (idx >= N_TOT*4*KK) return;
  int k = idx % KK;
  int nh = idx / KK;
  int h = nh & 3;
  const float* q = &qp[(size_t)nh*MP];
  const float* s = &kssum[(h*KK + k)*MP];
  float acc = 0.f;
#pragma unroll 6
  for (int m = 0; m < MM; ++m) acc += q[m]*s[m];
  zden[idx] = acc;
}

// -------- K6: zout[n,h,d] = (1/K) sum_k (qp @ kvs[h,k]) / zden ------------
__global__ __launch_bounds__(256) void k_out1(const __bf16* __restrict__ qp_bf,
    const __bf16* __restrict__ kvs_bf, const float* __restrict__ zden,
    float* __restrict__ zout)
{
  __shared__ float invd[64*40];
  const int tid = threadIdx.x, lane = tid & 31, wv = tid >> 5;
  const int n0 = blockIdx.x * 64;

  for (int i = tid; i < 64*40; i += 256){
    int n = n0 + i/40;
    float d = (n < N_TOT) ? zden[(size_t)n*40 + (i % 40)] : 1.f;
    invd[i] = 0.1f / d;                    // folds 1/K
  }
  __syncthreads();

  const int col16 = lane & 15;
  const int hi8  = (lane & 16) ? 8 : 0;

  for (int cb = wv; cb < 16; cb += 8){
    int rt = cb & 3, h = cb >> 2;
    // A: rows n0+rt*16.., stride 4*MP between consecutive n (tail zero-padded)
    v16bf a = load_a_vec(qp_bf + ((size_t)(n0 + rt*16)*4 + h)*MP, 4*MP, lane);

    v8f zero = {};
    v8f acc[4];
#pragma unroll
    for (int i = 0; i < 4; ++i) acc[i] = zero;

    for (int k = 0; k < KK; ++k){
#pragma unroll
      for (int dt = 0; dt < 4; ++dt){
        v16bf b = load_b_packed(kvs_bf + (size_t)((h*KK + k)*4 + dt)*512, lane);
        v8f t = wmma_bf16(a, b, zero);
#pragma unroll
        for (int r = 0; r < 8; ++r){
          int rowl = rt*16 + hi8 + r;
          acc[dt][r] += t[r] * invd[rowl*40 + h*KK + k];
        }
      }
    }
#pragma unroll
    for (int dt = 0; dt < 4; ++dt){
#pragma unroll
      for (int r = 0; r < 8; ++r){
        int n = n0 + rt*16 + hi8 + r;
        if (n < N_TOT) zout[(size_t)n*256 + h*64 + dt*16 + col16] = acc[dt][r];
      }
    }
  }
}

// ---------------- K7: out = zout @ Wo + bo --------------------------------
__global__ __launch_bounds__(256) void k_out2(const float* __restrict__ zout,
    const __bf16* __restrict__ wo_b, const float* __restrict__ bo,
    float* __restrict__ out)
{
  __shared__ __align__(32) __bf16 zs[64][256];
  const int tid = threadIdx.x, lane = tid & 31, wv = tid >> 5;
  const int n0 = blockIdx.x * 64;
  for (int i = tid; i < 64*64; i += 256){
    int r = i >> 6, c4 = (i & 63) * 4;
    int n = n0 + r;
    float4 zv = (n < N_TOT) ? *(const float4*)(zout + (size_t)n*256 + c4)
                            : make_float4(0.f, 0.f, 0.f, 0.f);
    zs[r][c4+0] = f2bf(zv.x); zs[r][c4+1] = f2bf(zv.y);
    zs[r][c4+2] = f2bf(zv.z); zs[r][c4+3] = f2bf(zv.w);
  }
  __syncthreads();
  const int col16 = lane & 15;
  const int rb = (lane & 16) ? 8 : 0;
  for (int t = wv; t < 16; t += 8){
    int rt = t >> 2, ct = t & 3;
    v8f c = {};
#pragma unroll
    for (int ks = 0; ks < 8; ++ks){
      v16bf a = load_a_vec(&zs[rt*16][0] + ks*32, 256, lane);
      v16bf b = load_b_packed(wo_b + (size_t)(ks*4 + ct)*512, lane);
      c = wmma_bf16(a, b, c);
    }
    int col = ct*16 + col16;
    float bias = bo[col];
#pragma unroll
    for (int r = 0; r < 8; ++r){
      int n = n0 + rt*16 + rb + r;
      if (n < N_TOT) out[(size_t)n*64 + col] = c[r] + bias;
    }
  }
}

extern "C" void kernel_launch(void* const* d_in, const int* in_sizes, int n_in,
                              void* d_out, int out_size, void* d_ws, size_t ws_size,
                              hipStream_t stream) {
  (void)in_sizes; (void)n_in; (void)out_size; (void)ws_size;
  const float* z    = (const float*)d_in[0];
  const float* Wq   = (const float*)d_in[1];
  const float* bq   = (const float*)d_in[2];
  const float* Wk   = (const float*)d_in[3];
  const float* bk   = (const float*)d_in[4];
  const float* Wv   = (const float*)d_in[5];
  const float* bv   = (const float*)d_in[6];
  const float* Wo   = (const float*)d_in[7];
  const float* bo   = (const float*)d_in[8];
  const float* proj = (const float*)d_in[9];
  const float* gum  = (const float*)d_in[10];
  const float* tau  = (const float*)d_in[11];
  float* out = (float*)d_out;

  char* ws = (char*)d_ws;
  size_t off = 0;
  auto take = [&](size_t bytes) -> char* {
    char* p = ws + off;
    off = (off + bytes + 255) & ~(size_t)255;
    return p;
  };
  __bf16* wcat   = (__bf16*)take((size_t)64*768*2);
  __bf16* wo_b   = (__bf16*)take((size_t)256*64*2);
  __bf16* projb  = (__bf16*)take((size_t)4*512*2);
  float*  vbuf   = (float*)take((size_t)N_TOT*256*4);
  float*  qp     = (float*)take((size_t)N_TOT*4*MP*4);
  __bf16* qp_bf  = (__bf16*)take((size_t)(N_TOT+64)*4*MP*2);  // tail zero-padded
  float*  kd     = (float*)take((size_t)N_TOT*4*MP*4);
  float*  diagk  = (float*)take((size_t)N_TOT*4*4);
  float*  pmax   = (float*)take((size_t)3125*4*4);
  float*  hmax   = (float*)take(64);
  float*  kvs    = (float*)take((size_t)4*KK*MP*64*4);
  __bf16* kvs_bf = (__bf16*)take((size_t)4*KK*MP*64*2);
  float*  kssum  = (float*)take((size_t)4*KK*MP*4);
  float*  zden   = (float*)take((size_t)N_TOT*4*KK*4);
  float*  zout   = (float*)take((size_t)N_TOT*256*4);

  __bf16* qp_bf_pad = qp_bf + (size_t)N_TOT*4*MP;

  k_prep<<<621, 256, 0, stream>>>(Wq, Wk, Wv, Wo, proj, wcat, wo_b,
                                  kvs, kssum, qp_bf_pad, projb);
  k_qkv<<<3125, 256, 0, stream>>>(z, wcat, projb, bq, bk, bv, tau,
                                  vbuf, qp, qp_bf, kd, diagk, pmax);
  k_hmax<<<4, 256, 0, stream>>>(pmax, hmax, 3125);
  k_kp<<<25000, 256, 0, stream>>>(kd, diagk, hmax);
  dim3 gkvs(98, 4);
  k_kvs<<<gkvs, 256, 0, stream>>>(kd, vbuf, gum, tau, kvs, kssum);
  k_cvt<<<320, 256, 0, stream>>>(kvs, kvs_bf);
  k_zden<<<(N_TOT*4*KK + 255)/256, 256, 0, stream>>>(qp, kssum, zden);
  k_out1<<<782, 256, 0, stream>>>(qp_bf, kvs_bf, zden, zout);
  k_out2<<<782, 256, 0, stream>>>(zout, wo_b, bo, out);
}
